// SemanticModule_5935644803438
// MI455X (gfx1250) — matrix operand order repaired and checked by reading, not compiled, not run on hardware
//
#include <hip/hip_runtime.h>

// ---------------------------------------------------------------------------
// CDNA5 / gfx1250, wave32, v_wmma_f32_16x16x32_bf16.
// - Aggregate-before-transform for add/mean relations (node tables fit the
//   192MB L2, so the scatter-adds are L2 atomics, not HBM).
// - Max relation: per-edge WMMA GEMM + monotone-encoded u32 atomic-max
//   (branch-free: OOB edges atomically max with 0u == guaranteed no-op).
// - Pre-transposed bf16 weights: B fragment = one contiguous 32B load.
// - GEMM kernels templated on <DIN,DOUT,PROJ>: K-loops fully unroll.
// ---------------------------------------------------------------------------

typedef __attribute__((ext_vector_type(16))) __bf16 v16bf;
typedef __attribute__((ext_vector_type(8)))  float  v8f;
typedef __attribute__((ext_vector_type(8)))  float  f8v;

#define SENT_MAX 0x007FFFFFu  // monotone encoding of -inf (minimum encoding)

__device__ __forceinline__ unsigned fenc(float x) {
  unsigned u = __float_as_uint(x);
  return (u & 0x80000000u) ? ~u : (u | 0x80000000u);
}
__device__ __forceinline__ float fdec(unsigned u) {
  return __uint_as_float((u & 0x80000000u) ? (u & 0x7FFFFFFFu) : ~u);
}

// A-fragment (16x32 bf16): row m = lane&15; lanes<16 hold K 0..15 +k0,
// lanes>=16 hold K 16..31 +k0.
// B-fragment (32x16): col n = lane&15; lanes<16 K 0..15, lanes>=16 K 16..31.

// One (A-source, weight) pair accumulated into 4 adjacent 16x16 n-tiles.
template <int DIN>
__device__ __forceinline__ void gemm_pair(const float* __restrict__ rowp,
                                          float s,
                                          const __bf16* __restrict__ Wt,
                                          int nb0, v8f acc[4]) {
  constexpr int KPAD = (DIN + 31) & ~31;
  constexpr int NK = KPAD / 32;
  const int lane = threadIdx.x & 31;
  const int hi8 = (lane >> 4) << 4;
  const int ncol = lane & 15;
  const __bf16* wbase = Wt + (size_t)(nb0 + ncol) * KPAD + ((lane >> 4) << 4);
#pragma unroll
  for (int kk = 0; kk < NK; ++kk) {
    const int k0 = kk * 32;
    const int kb = k0 + hi8;
    v16bf a;
    if constexpr (DIN % 32 == 0) {  // full tiles: 2 aligned 32B vector loads
      f8v x0 = *(const f8v*)(rowp + kb);
      f8v x1 = *(const f8v*)(rowp + kb + 8);
#pragma unroll
      for (int i = 0; i < 8; ++i) {
        a[i]     = (__bf16)(x0[i] * s);
        a[i + 8] = (__bf16)(x1[i] * s);
      }
    } else {  // DIN=6 first layer: guarded scalar loads, zero-padded
#pragma unroll
      for (int i = 0; i < 16; ++i) {
        int k = kb + i;
        float v = (k < DIN) ? rowp[k] * s : 0.0f;
        a[i] = (__bf16)v;
      }
    }
#pragma unroll
    for (int t = 0; t < 4; ++t) {
      v16bf b = *(const v16bf*)(wbase + (size_t)t * 16 * KPAD + k0);
      acc[t] = __builtin_amdgcn_wmma_f32_16x16x32_bf16(false, a, false, b,
                                                       (short)0, acc[t],
                                                       false, false);
    }
  }
}

// ---------------------------------------------------------------------------
// weight prep: Wt[n*Kpad + k] = bf16(W1[k*N+n] (+ W2[k*N+n])), zero-pad K
__global__ void k_prep_w(const float* __restrict__ W1,
                         const float* __restrict__ W2, int K, int N, int Kpad,
                         __bf16* __restrict__ Wt) {
  long long total = (long long)N * Kpad;
  long long i = blockIdx.x * (long long)blockDim.x + threadIdx.x;
  long long stride = (long long)gridDim.x * blockDim.x;
  for (; i < total; i += stride) {
    int n = (int)(i / Kpad);
    int k = (int)(i - (long long)n * Kpad);
    float v = 0.0f;
    if (k < K) {
      v = W1[(size_t)k * N + n];
      if (W2) v += W2[(size_t)k * N + n];
    }
    Wt[i] = (__bf16)v;
  }
}

__global__ void k_count(const int* __restrict__ dst, long long E,
                        float* __restrict__ cnt) {
  long long i = blockIdx.x * (long long)blockDim.x + threadIdx.x;
  long long stride = (long long)gridDim.x * blockDim.x;
  for (; i < E; i += stride) atomicAdd(&cnt[dst[i]], 1.0f);
}

__global__ void k_fill_u32(unsigned* __restrict__ p, long long n, unsigned v) {
  long long i = blockIdx.x * (long long)blockDim.x + threadIdx.x;
  long long stride = (long long)gridDim.x * blockDim.x;
  for (; i < n; i += stride) p[i] = v;
}

// scatter-add of raw source features (shift/mask when din is a power of two)
__global__ void k_scatter_add(const float* __restrict__ X, int ldx,
                              const int* __restrict__ src,
                              const int* __restrict__ dst, long long E,
                              int din, int shift, float* __restrict__ agg) {
  long long total = E * din;
  long long i = blockIdx.x * (long long)blockDim.x + threadIdx.x;
  long long stride = (long long)gridDim.x * blockDim.x;
  for (; i < total; i += stride) {
    long long e;
    int f;
    if (shift >= 0) {
      e = i >> shift;
      f = (int)i & (din - 1);
    } else {
      e = i / din;
      f = (int)(i - e * din);
    }
    atomicAdd(&agg[(size_t)dst[e] * din + f], X[(size_t)src[e] * ldx + f]);
  }
}

// per-edge WMMA GEMM (msg = x[src] @ Wm) + atomic segment-max scatter.
// One wave: 16 edges x 64 output columns. Branch-free epilogue: OOB edges
// write value 0u (always <= SENT_MAX-initialized buffer -> no-op).
template <int DIN, int DOUT>
__global__ __launch_bounds__(32)
void k_edge_gemm_max(const float* __restrict__ X, int ldx,
                     const __bf16* __restrict__ Wt,
                     const int* __restrict__ src, const int* __restrict__ dst,
                     int E, unsigned* __restrict__ aggmax) {
  int e0 = blockIdx.x << 4;
  int nb0 = blockIdx.y << 6;
  int lane = threadIdx.x & 31;
  int hi8 = (lane >> 4) << 3;
  int ee = e0 + (lane & 15);
  if (ee >= E) ee = E - 1;  // clamp: valid address, result never stored
  const float* rowp = X + (size_t)src[ee] * ldx;

  // this lane's 8 destination node ids (consecutive edges), loaded once
  int ebase = e0 + hi8;
  int d[8];
  if (ebase + 8 <= E) {
    int4 d0 = *(const int4*)(dst + ebase);
    int4 d1 = *(const int4*)(dst + ebase + 4);
    d[0] = d0.x; d[1] = d0.y; d[2] = d0.z; d[3] = d0.w;
    d[4] = d1.x; d[5] = d1.y; d[6] = d1.z; d[7] = d1.w;
  } else {
#pragma unroll
    for (int r = 0; r < 8; ++r) {
      int e = ebase + r;
      d[r] = dst[e < E ? e : E - 1];
    }
  }

  v8f acc[4];
#pragma unroll
  for (int t = 0; t < 4; ++t) acc[t] = (v8f){0.f,0.f,0.f,0.f,0.f,0.f,0.f,0.f};

  gemm_pair<DIN>(rowp, 1.0f, Wt, nb0, acc);

#pragma unroll
  for (int t = 0; t < 4; ++t) {
    int col = nb0 + t * 16 + (lane & 15);
#pragma unroll
    for (int r = 0; r < 8; ++r) {
      unsigned val = (ebase + r < E) ? fenc(acc[t][r]) : 0u;  // 0u -> no-op
      atomicMax(&aggmax[(size_t)d[r] * DOUT + col], val);
    }
  }
}

// fused node conv: acc = X@Wt0 + (A1/cnt1)@Wt1 + (A2/cnt2)@Wt2
// out = relu(acc + b1 + b2 + decoded_max) + (resX@WtR | resAdd)
// One wave: 16 rows x 64 output columns.
template <int DIN, int DOUT, bool PROJ>
__global__ __launch_bounds__(32)
void k_node_gemm(const float* __restrict__ X, int ldx,
                 const __bf16* __restrict__ Wt0,
                 const float* __restrict__ A1, const float* __restrict__ cnt1,
                 const __bf16* __restrict__ Wt1,
                 const float* __restrict__ A2, const float* __restrict__ cnt2,
                 const __bf16* __restrict__ Wt2,
                 const float* __restrict__ b1, const float* __restrict__ b2,
                 const unsigned* __restrict__ maxenc,
                 const float* __restrict__ resX, int ldres,
                 const __bf16* __restrict__ WtR,
                 const float* __restrict__ resAdd, int ldadd,
                 float* __restrict__ out, int M, int do_relu) {
  int row0 = blockIdx.x << 4;
  int nb0 = blockIdx.y << 6;
  int lane = threadIdx.x & 31;
  int hi8 = (lane >> 4) << 3;
  int mrow = row0 + (lane & 15);
  if (mrow >= M) mrow = M - 1;  // clamp: valid address, never stored

  v8f acc[4];
#pragma unroll
  for (int t = 0; t < 4; ++t) acc[t] = (v8f){0.f,0.f,0.f,0.f,0.f,0.f,0.f,0.f};

  gemm_pair<DIN>(X + (size_t)mrow * ldx, 1.0f, Wt0, nb0, acc);
  if (A1) {
    float s = cnt1 ? (1.0f / fmaxf(cnt1[mrow], 1.0f)) : 1.0f;
    gemm_pair<DIN>(A1 + (size_t)mrow * DIN, s, Wt1, nb0, acc);
  }
  if (A2) {
    float s = cnt2 ? (1.0f / fmaxf(cnt2[mrow], 1.0f)) : 1.0f;
    gemm_pair<DIN>(A2 + (size_t)mrow * DIN, s, Wt2, nb0, acc);
  }

  v8f accR[4];
#pragma unroll
  for (int t = 0; t < 4; ++t) accR[t] = (v8f){0.f,0.f,0.f,0.f,0.f,0.f,0.f,0.f};
  if constexpr (PROJ) {  // layer-1 projected residual: resX[M x 64] @ WtR
    gemm_pair<64>(resX + (size_t)mrow * ldres, 1.0f, WtR, nb0, accR);
  }

  bool full = (row0 + 16 <= M);
#pragma unroll
  for (int t = 0; t < 4; ++t) {
    int col = nb0 + t * 16 + (lane & 15);
    float bias = (b1 ? b1[col] : 0.0f) + (b2 ? b2[col] : 0.0f);
    if (full) {  // whole-wave fast path: no per-element row guards
#pragma unroll
      for (int r = 0; r < 8; ++r) {
        int row = row0 + r + hi8;
        float v = acc[t][r] + bias;
        if (maxenc) {
          unsigned u = maxenc[(size_t)row * DOUT + col];
          if (u != SENT_MAX) v += fdec(u);
        }
        if (do_relu) v = fmaxf(v, 0.0f);
        if constexpr (PROJ) v += accR[t][r];
        if (resAdd) v += resAdd[(size_t)row * ldadd + col];
        out[(size_t)row * DOUT + col] = v;
      }
    } else {
#pragma unroll
      for (int r = 0; r < 8; ++r) {
        int row = row0 + r + hi8;
        if (row < M) {
          float v = acc[t][r] + bias;
          if (maxenc) {
            unsigned u = maxenc[(size_t)row * DOUT + col];
            if (u != SENT_MAX) v += fdec(u);
          }
          if (do_relu) v = fmaxf(v, 0.0f);
          if constexpr (PROJ) v += accR[t][r];
          if (resAdd) v += resAdd[(size_t)row * ldadd + col];
          out[(size_t)row * DOUT + col] = v;
        }
      }
    }
  }
}

// ---------------------------------------------------------------------------
static inline int gblocks(long long total, int tpb, int cap) {
  long long b = (total + tpb - 1) / tpb;
  if (b > cap) b = cap;
  if (b < 1) b = 1;
  return (int)b;
}

extern "C" void kernel_launch(void* const* d_in, const int* in_sizes, int n_in,
                              void* d_out, int out_size, void* d_ws,
                              size_t ws_size, hipStream_t stream) {
  const int NS = 100000, NB = 50000;
  // REL order: 0 temp_previous(add,s->s) 1 intersects(mean,s->s)
  //            2 represented_by(mean,s->b) 3 brepcoplanar(max,b->b)
  const float *wm[5][4], *wr[5][4], *bb[5][4];
  const float *proj_s, *proj_b, *final_w, *final_b, *x_s, *x_b;
  const int *e_tp, *e_in, *e_rb, *e_bc;
  long long E_tp, E_in, E_rb, E_bc;

  if (in_sizes[0] == NS * 6) {
    // python-dict insertion-order flatten
    x_s = (const float*)d_in[0];
    x_b = (const float*)d_in[1];
    int idx = 2;
    for (int l = 0; l < 5; ++l)
      for (int r = 0; r < 4; ++r) {
        wm[l][r] = (const float*)d_in[idx++];
        wr[l][r] = (const float*)d_in[idx++];
        bb[l][r] = (const float*)d_in[idx++];
      }
    proj_s  = (const float*)d_in[62];
    proj_b  = (const float*)d_in[63];
    final_w = (const float*)d_in[64];
    final_b = (const float*)d_in[65];
    e_tp = (const int*)d_in[66]; E_tp = in_sizes[66] / 2;
    e_in = (const int*)d_in[67]; E_in = in_sizes[67] / 2;
    e_rb = (const int*)d_in[68]; E_rb = in_sizes[68] / 2;
    e_bc = (const int*)d_in[69]; E_bc = in_sizes[69] / 2;
  } else {
    // jax pytree (sorted-key) flatten
    e_bc = (const int*)d_in[0]; E_bc = in_sizes[0] / 2;
    e_in = (const int*)d_in[1]; E_in = in_sizes[1] / 2;
    e_rb = (const int*)d_in[2]; E_rb = in_sizes[2] / 2;
    e_tp = (const int*)d_in[3]; E_tp = in_sizes[3] / 2;
    final_b = (const float*)d_in[4];
    final_w = (const float*)d_in[5];
    int idx = 6;
    const int relmap[4] = {3, 1, 2, 0};
    for (int l = 0; l < 5; ++l)
      for (int j = 0; j < 4; ++j) {
        int r = relmap[j];
        bb[l][r] = (const float*)d_in[idx++];  // keys sorted: b, w_msg, w_root
        wm[l][r] = (const float*)d_in[idx++];
        wr[l][r] = (const float*)d_in[idx++];
      }
    proj_b = (const float*)d_in[66];
    proj_s = (const float*)d_in[67];
    x_b    = (const float*)d_in[68];
    x_s    = (const float*)d_in[69];
  }
  (void)n_in; (void)ws_size; (void)out_size;

  // workspace layout (256B aligned blocks)
  char* wptr = (char*)d_ws;
  auto alloc = [&](size_t bytes) {
    void* p = (void*)wptr;
    wptr += (bytes + 255) & ~(size_t)255;
    return p;
  };
  float* hs[2]; float* hb[2];
  hs[0] = (float*)alloc((size_t)NS * 128 * 4);
  hs[1] = (float*)alloc((size_t)NS * 128 * 4);
  hb[0] = (float*)alloc((size_t)NB * 128 * 4);
  hb[1] = (float*)alloc((size_t)NB * 128 * 4);
  float*    agg_tp = (float*)alloc((size_t)NS * 128 * 4);
  float*    agg_in = (float*)alloc((size_t)NS * 128 * 4);
  float*    agg_rb = (float*)alloc((size_t)NB * 128 * 4);
  unsigned* aggmax = (unsigned*)alloc((size_t)NB * 128 * 4);
  float*    cnt_in = (float*)alloc((size_t)NS * 4);
  float*    cnt_rb = (float*)alloc((size_t)NB * 4);
  // pre-transposed bf16 weights, [dout x Kpad], reused across layers
  __bf16* wt_root_s = (__bf16*)alloc((size_t)128 * 128 * 2);
  __bf16* wt_tp     = (__bf16*)alloc((size_t)128 * 128 * 2);
  __bf16* wt_in     = (__bf16*)alloc((size_t)128 * 128 * 2);
  __bf16* wt_root_b = (__bf16*)alloc((size_t)128 * 128 * 2);
  __bf16* wt_rb     = (__bf16*)alloc((size_t)128 * 128 * 2);
  __bf16* wt_bc     = (__bf16*)alloc((size_t)128 * 128 * 2);
  __bf16* wt_proj_s = (__bf16*)alloc((size_t)128 * 64 * 2);
  __bf16* wt_proj_b = (__bf16*)alloc((size_t)128 * 64 * 2);
  __bf16* wt_final  = (__bf16*)alloc((size_t)128 * 128 * 2);

  // destination counts (edges static -> once per launch)
  (void)hipMemsetAsync(cnt_in, 0, (size_t)NS * 4, stream);
  (void)hipMemsetAsync(cnt_rb, 0, (size_t)NB * 4, stream);
  k_count<<<gblocks(E_in, 256, 65535), 256, 0, stream>>>(e_in + E_in, E_in, cnt_in);
  k_count<<<gblocks(E_rb, 256, 65535), 256, 0, stream>>>(e_rb + E_rb, E_rb, cnt_rb);

  // proj / final weight prep (buffers not reused elsewhere)
  k_prep_w<<<gblocks(128 * 64, 256, 65535), 256, 0, stream>>>(proj_s, nullptr, 64, 128, 64, wt_proj_s);
  k_prep_w<<<gblocks(128 * 64, 256, 65535), 256, 0, stream>>>(proj_b, nullptr, 64, 128, 64, wt_proj_b);
  k_prep_w<<<gblocks(128 * 128, 256, 65535), 256, 0, stream>>>(final_w, nullptr, 128, 128, 128, wt_final);

  const int din_a[5]  = {6, 64, 128, 128, 128};
  const int dout_a[5] = {64, 128, 128, 128, 128};
  const float* xs_s = x_s; int ld_s = 6;
  const float* xs_b = x_b; int ld_b = 6;

  for (int l = 0; l < 5; ++l) {
    int din = din_a[l], dout = dout_a[l];
    int Kpad = (din + 31) & ~31;
    int shift = (din == 64) ? 6 : (din == 128) ? 7 : -1;

    // weight prep for this layer (stream-serial: safe to reuse buffers)
    long long wtn = (long long)dout * Kpad;
    k_prep_w<<<gblocks(wtn, 256, 65535), 256, 0, stream>>>(wr[l][0], wr[l][1], din, dout, Kpad, wt_root_s);
    k_prep_w<<<gblocks(wtn, 256, 65535), 256, 0, stream>>>(wm[l][0], nullptr,  din, dout, Kpad, wt_tp);
    k_prep_w<<<gblocks(wtn, 256, 65535), 256, 0, stream>>>(wm[l][1], nullptr,  din, dout, Kpad, wt_in);
    k_prep_w<<<gblocks(wtn, 256, 65535), 256, 0, stream>>>(wr[l][2], wr[l][3], din, dout, Kpad, wt_root_b);
    k_prep_w<<<gblocks(wtn, 256, 65535), 256, 0, stream>>>(wm[l][2], nullptr,  din, dout, Kpad, wt_rb);
    k_prep_w<<<gblocks(wtn, 256, 65535), 256, 0, stream>>>(wm[l][3], nullptr,  din, dout, Kpad, wt_bc);

    (void)hipMemsetAsync(agg_tp, 0, (size_t)NS * din * 4, stream);
    (void)hipMemsetAsync(agg_in, 0, (size_t)NS * din * 4, stream);
    (void)hipMemsetAsync(agg_rb, 0, (size_t)NB * din * 4, stream);
    k_fill_u32<<<gblocks((long long)NB * dout, 256, 65535), 256, 0, stream>>>(
        aggmax, (long long)NB * dout, SENT_MAX);

    // aggregate-first for linear (add/mean) aggregations
    k_scatter_add<<<gblocks(E_tp * din, 256, 262144), 256, 0, stream>>>(
        xs_s, ld_s, e_tp, e_tp + E_tp, E_tp, din, shift, agg_tp);
    k_scatter_add<<<gblocks(E_in * din, 256, 262144), 256, 0, stream>>>(
        xs_s, ld_s, e_in, e_in + E_in, E_in, din, shift, agg_in);
    k_scatter_add<<<gblocks(E_rb * din, 256, 262144), 256, 0, stream>>>(
        xs_s, ld_s, e_rb, e_rb + E_rb, E_rb, din, shift, agg_rb);

    // max aggregation: per-edge WMMA transform + atomic-max scatter
    dim3 ge((unsigned)((E_bc + 15) / 16), (unsigned)(dout / 64));
    if (l == 0) {
      k_edge_gemm_max<6, 64><<<ge, 32, 0, stream>>>(xs_b, ld_b, wt_bc, e_bc, e_bc + E_bc, (int)E_bc, aggmax);
    } else if (l == 1) {
      k_edge_gemm_max<64, 128><<<ge, 32, 0, stream>>>(xs_b, ld_b, wt_bc, e_bc, e_bc + E_bc, (int)E_bc, aggmax);
    } else {
      k_edge_gemm_max<128, 128><<<ge, 32, 0, stream>>>(xs_b, ld_b, wt_bc, e_bc, e_bc + E_bc, (int)E_bc, aggmax);
    }

    float* out_s = hs[l & 1];
    float* out_b = hb[l & 1];
    const float *resAdd_s = nullptr, *resAdd_b = nullptr;
    int ldadd_s = 0, ldadd_b = 0;
    if (l >= 2) {
      resAdd_s = xs_s; ldadd_s = ld_s;
      resAdd_b = xs_b; ldadd_b = ld_b;
    }

    dim3 gs((NS + 15) / 16, dout / 64);
    dim3 gb((NB + 15) / 16, dout / 64);
    if (l == 0) {
      k_node_gemm<6, 64, false><<<gs, 32, 0, stream>>>(
          xs_s, ld_s, wt_root_s, agg_tp, nullptr, wt_tp, agg_in, cnt_in, wt_in,
          bb[l][0], bb[l][1], nullptr, nullptr, 0, nullptr, nullptr, 0,
          out_s, NS, 1);
      k_node_gemm<6, 64, false><<<gb, 32, 0, stream>>>(
          xs_b, ld_b, wt_root_b, agg_rb, cnt_rb, wt_rb, nullptr, nullptr, nullptr,
          bb[l][2], bb[l][3], aggmax, nullptr, 0, nullptr, nullptr, 0,
          out_b, NB, 1);
    } else if (l == 1) {
      k_node_gemm<64, 128, true><<<gs, 32, 0, stream>>>(
          xs_s, ld_s, wt_root_s, agg_tp, nullptr, wt_tp, agg_in, cnt_in, wt_in,
          bb[l][0], bb[l][1], nullptr, xs_s, ld_s, wt_proj_s, nullptr, 0,
          out_s, NS, 1);
      k_node_gemm<64, 128, true><<<gb, 32, 0, stream>>>(
          xs_b, ld_b, wt_root_b, agg_rb, cnt_rb, wt_rb, nullptr, nullptr, nullptr,
          bb[l][2], bb[l][3], aggmax, xs_b, ld_b, wt_proj_b, nullptr, 0,
          out_b, NB, 1);
    } else {
      k_node_gemm<128, 128, false><<<gs, 32, 0, stream>>>(
          xs_s, ld_s, wt_root_s, agg_tp, nullptr, wt_tp, agg_in, cnt_in, wt_in,
          bb[l][0], bb[l][1], nullptr, nullptr, 0, nullptr, resAdd_s, ldadd_s,
          out_s, NS, 1);
      k_node_gemm<128, 128, false><<<gb, 32, 0, stream>>>(
          xs_b, ld_b, wt_root_b, agg_rb, cnt_rb, wt_rb, nullptr, nullptr, nullptr,
          bb[l][2], bb[l][3], aggmax, nullptr, 0, nullptr, resAdd_b, ldadd_b,
          out_b, NB, 1);
    }

    xs_s = out_s; ld_s = dout;
    xs_b = out_b; ld_b = dout;
  }

  // final projection: d_out = xs_b @ final_w + final_b (no relu)
  dim3 gf((NB + 15) / 16, 128 / 64);
  k_node_gemm<128, 128, false><<<gf, 32, 0, stream>>>(
      xs_b, ld_b, wt_final,
      nullptr, nullptr, nullptr,
      nullptr, nullptr, nullptr,
      final_b, nullptr, nullptr,
      nullptr, 0, nullptr, nullptr, 0,
      (float*)d_out, NB, 0);
}
